// HIL_70961449664962
// MI455X (gfx1250) — matrix operand order repaired and checked by reading, not compiled, not run on hardware
//
#include <hip/hip_runtime.h>

// ---------------------------------------------------------------------------
// GNN message passing (3 layers, D=128) for gfx1250 using bf16 WMMA + TDM
// ---------------------------------------------------------------------------

#define DEV __device__ __forceinline__

typedef __attribute__((ext_vector_type(16))) __bf16 v16bf;
typedef __attribute__((ext_vector_type(8)))  float  v8f;
typedef __attribute__((ext_vector_type(4)))  unsigned int v4u;
typedef __attribute__((ext_vector_type(8)))  int  v8i;
typedef __attribute__((ext_vector_type(4)))  int  v4i;

constexpr int Nn  = 50000;
constexpr int Ee  = 600000;
constexpr int Dd  = 128;
constexpr int ECc = 16;
constexpr int Ll  = 3;
constexpr float CUT = 10.0f;

// Packed weight layout (in bf16 "halfs"):
//   fragment = 32 lanes x 16 halfs (one WMMA B operand, 32x16 K-major tile)
//   128x128 matrix = 4 k-tiles x 8 n-tiles = 32 fragments
// Per layer, order: Wsrc, Wdst, Wmsg, Wself, Wagg, Wedge(16x128 padded to 32)
constexpr int FRAG        = 32 * 16;          // 512 halfs
constexpr int MAT128      = 4 * 8 * FRAG;     // 16384 halfs
constexpr int MATE        = 8 * FRAG;         // 4096 halfs
constexpr int LAYER_HALFS = 5 * MAT128 + MATE; // 86016 halfs

DEV float silu_f(float x) { return x / (1.0f + __expf(-x)); }

// WMMA: D = A(16x32 bf16) * B(32x16 bf16) + C(16x16 f32)
DEV v8f wmma_bf16(v16bf a, v16bf b, v8f c) {
    return __builtin_amdgcn_wmma_f32_16x16x32_bf16(
        /*neg_a=*/false, a, /*neg_b=*/false, b,
        /*c_mod=*/(short)0, c, /*reuse_a=*/false, /*reuse_b=*/false);
}

// Load one B fragment (packed order) from global or LDS.
DEV v16bf ldfrag(const __bf16* base, int kt, int nt, int lane) {
    return *(const v16bf*)(base + ((size_t)((kt * 8 + nt) * 32 + lane)) * 16);
}

// Load one A fragment (16x32 slice starting at column k0) from an fp32 row
// pointer, converting to bf16.  A layout (16-bit, 16x32):
//   lane<16 : row=lane,    halfs j<8 -> K=k0+j,    j>=8 -> K=k0+16+(j-8)
//   lane>=16: row=lane-16, halfs j<8 -> K=k0+8+j,  j>=8 -> K=k0+24+(j-8)
DEV v16bf load_a(const float* row, int k0, int lane) {
    int c = (lane & 16) ? 8 : 0;
    const float* p = row + k0 + c;
    float4 f0 = *(const float4*)(p);
    float4 f1 = *(const float4*)(p + 4);
    float4 f2 = *(const float4*)(p + 16);
    float4 f3 = *(const float4*)(p + 20);
    v16bf a;
    a[0]  = (__bf16)f0.x; a[1]  = (__bf16)f0.y; a[2]  = (__bf16)f0.z; a[3]  = (__bf16)f0.w;
    a[4]  = (__bf16)f1.x; a[5]  = (__bf16)f1.y; a[6]  = (__bf16)f1.z; a[7]  = (__bf16)f1.w;
    a[8]  = (__bf16)f2.x; a[9]  = (__bf16)f2.y; a[10] = (__bf16)f2.z; a[11] = (__bf16)f2.w;
    a[12] = (__bf16)f3.x; a[13] = (__bf16)f3.y; a[14] = (__bf16)f3.z; a[15] = (__bf16)f3.w;
    return a;
}

// ---------------------------------------------------------------------------
// Weight packing: fp32 row-major [K][N] -> bf16 B-fragment order.
// B layout (16-bit, 32x16): lane<16 holds col=lane, j<8 -> K=kt*32+j, j>=8 -> K=kt*32+16+(j-8);
//                           lane>=16 holds col=lane-16, j<8 -> K=kt*32+8+j, j>=8 -> K=kt*32+24+(j-8).
// ---------------------------------------------------------------------------
__global__ void pack_kernel(const float* __restrict__ Wsrc,
                            const float* __restrict__ Wdst,
                            const float* __restrict__ Wmsg,
                            const float* __restrict__ Wself,
                            const float* __restrict__ Wagg,
                            const float* __restrict__ Wedge,
                            __bf16* __restrict__ packed) {
    int idx = blockIdx.x * blockDim.x + threadIdx.x;
    int total = Ll * LAYER_HALFS;
    if (idx >= total) return;
    int layer = idx / LAYER_HALFS;
    int rem   = idx % LAYER_HALFS;

    const float* W;
    int K, kt, nt, within;
    if (rem < 5 * MAT128) {
        int mat = rem / MAT128;
        int off = rem % MAT128;
        const float* bases[5] = {Wsrc, Wdst, Wmsg, Wself, Wagg};
        W = bases[mat] + (size_t)layer * Dd * Dd;
        K = Dd;
        int frag = off / FRAG;
        kt = frag >> 3; nt = frag & 7;
        within = off % FRAG;
    } else {
        int off = rem - 5 * MAT128;
        W = Wedge + (size_t)layer * ECc * Dd;
        K = ECc;
        int frag = off / FRAG;
        kt = 0; nt = frag & 7;
        within = off % FRAG;
    }
    int lane = within / 16;
    int j    = within % 16;
    int k = kt * 32 + (j & 7) + ((lane & 16) ? 8 : 0) + ((j & 8) ? 16 : 0);
    int n = nt * 16 + (lane & 15);
    float v = (k < K) ? W[(size_t)k * Dd + n] : 0.0f;
    packed[idx] = (__bf16)v;
}

// ---------------------------------------------------------------------------
__global__ void cutoff_kernel(const float* __restrict__ dist,
                              float* __restrict__ Cenv) {
    int i = blockIdx.x * blockDim.x + threadIdx.x;
    if (i < Ee) {
        float d = dist[i];
        float c = 0.5f * (cosf(3.14159265358979323846f * d / CUT) + 1.0f);
        Cenv[i] = (d < CUT) ? c : 0.0f;
    }
}

__global__ void zero_kernel(float* __restrict__ p, int n) {
    int i = blockIdx.x * blockDim.x + threadIdx.x;
    if (i < n) p[i] = 0.0f;
}

// ---------------------------------------------------------------------------
// Edge kernel: per 16-edge wave-tile
//   pre  = b_edge + ef*We + h[src]*Ws + h[dst]*Wd        (72 WMMAs)
//   msg  = (silu(pre) * Wmsg) * C(dist)                  (32 WMMAs)
//   agg[dst] += msg                                      (atomics, L2-resident)
// W_msg fragments are DMA'd into LDS by the Tensor Data Mover.
// ---------------------------------------------------------------------------
__global__ __launch_bounds__(128)
void edge_kernel(const float* __restrict__ h,
                 const float* __restrict__ ef,
                 const int*   __restrict__ src,
                 const int*   __restrict__ dstv,
                 const float* __restrict__ Cenv,
                 const float* __restrict__ bedge,
                 const __bf16* __restrict__ pWsrc,
                 const __bf16* __restrict__ pWdst,
                 const __bf16* __restrict__ pWmsg,
                 const __bf16* __restrict__ pWedge,
                 float* __restrict__ agg) {
    __shared__ __bf16 sWmsg[MAT128];        // 32 KB: W_msg fragments
    __shared__ __bf16 stage[4][4 * FRAG];   // 16 KB: per-wave silu(pre) tile

    const int lane = threadIdx.x & 31;
    const int wave = threadIdx.x >> 5;

    // --- TDM: DMA W_msg fragment block (32 KB) straight into LDS -----------
    // D# group0: count=1, lds_addr, global_addr, type=2 ("image")
    // D# group1: data_size=1 (2B), tensor_dim0 = tile_dim0 = MAT128 elements,
    //            tensor_dim1 = 1, tensor_dim0_stride = MAT128.
    if (wave == 0) {
        unsigned lds_off = (unsigned)(unsigned long long)(const void*)sWmsg;
        unsigned long long ga = (unsigned long long)(const void*)pWmsg;
        v4u g0;
        g0[0] = 1u;                                           // count=1
        g0[1] = lds_off;                                      // lds_addr
        g0[2] = (unsigned)(ga & 0xffffffffull);               // global_addr lo
        g0[3] = (unsigned)((ga >> 32) & 0x1ffffffull)         // global_addr hi
                | (2u << 30);                                 // type=2
        v8i g1;
        g1[0] = (int)(1u << 16);                              // data_size=1 (2B)
        g1[1] = (int)(((unsigned)MAT128 & 0xffffu) << 16);    // tensor_dim0 lo16
        g1[2] = (int)(((unsigned)MAT128 >> 16) | (1u << 16)); // td0 hi16 | td1 lo16
        g1[3] = (int)(((unsigned)MAT128) << 16);              // td1 hi=0 | tile_dim0
        g1[4] = 0;                                            // tile_dim1/2 unused
        g1[5] = (int)MAT128;                                  // td0_stride lo32
        g1[6] = 0;
        g1[7] = 0;
        v4i zz4 = {0, 0, 0, 0};
        v8i zz8 = {0, 0, 0, 0, 0, 0, 0, 0};
        __builtin_amdgcn_tensor_load_to_lds(g0, g1, zz4, zz4, zz8, 0);
        __builtin_amdgcn_s_wait_tensorcnt(0);
    }
    __syncthreads();

    const int m16  = lane & 15;
    const int c8   = (lane & 16) ? 8 : 0;
    const int ntile = Ee / 16;   // 37500, exact

    for (int t = blockIdx.x * 4 + wave; t < ntile; t += gridDim.x * 4) {
        const int e0 = t * 16;

        // --- accumulators initialized with bias (row-invariant per column) ---
        v8f acc[8];
#pragma unroll
        for (int nt = 0; nt < 8; ++nt) {
            float b = bedge[nt * 16 + m16];
#pragma unroll
            for (int r = 0; r < 8; ++r) acc[nt][r] = b;
        }

        // --- edge-feature term (EC=16, zero-padded K tile) ---
        {
            v16bf aE;
            const float* er = ef + (size_t)(e0 + m16) * ECc + c8;
            float4 f0 = *(const float4*)er;
            float4 f1 = *(const float4*)(er + 4);
            aE[0] = (__bf16)f0.x; aE[1] = (__bf16)f0.y; aE[2] = (__bf16)f0.z; aE[3] = (__bf16)f0.w;
            aE[4] = (__bf16)f1.x; aE[5] = (__bf16)f1.y; aE[6] = (__bf16)f1.z; aE[7] = (__bf16)f1.w;
#pragma unroll
            for (int j = 8; j < 16; ++j) aE[j] = (__bf16)0.0f;
#pragma unroll
            for (int nt = 0; nt < 8; ++nt)
                acc[nt] = wmma_bf16(aE, ldfrag(pWedge, 0, nt, lane), acc[nt]);
        }

        // --- gathered endpoint GEMMs (h fits in L2 -> cheap gathers) ---
        const float* rs = h + (size_t)src[e0 + m16] * Dd;
        const float* rd = h + (size_t)dstv[e0 + m16] * Dd;
#pragma unroll
        for (int kt = 0; kt < 4; ++kt) {
            v16bf as = load_a(rs, kt * 32, lane);
            v16bf ad = load_a(rd, kt * 32, lane);
#pragma unroll
            for (int nt = 0; nt < 8; ++nt) {
                acc[nt] = wmma_bf16(as, ldfrag(pWsrc, kt, nt, lane), acc[nt]);
                acc[nt] = wmma_bf16(ad, ldfrag(pWdst, kt, nt, lane), acc[nt]);
            }
        }

        // --- silu, reshuffle C-layout -> A-layout via per-wave LDS stage ---
        __bf16* st = stage[wave];
#pragma unroll
        for (int nt = 0; nt < 8; ++nt) {
            int kk  = nt * 16 + m16;        // column index == K index of GEMM2
            int kt2 = kk >> 5;
            int kl  = kk & 31;
            int hi  = (kl & 8) ? 16 : 0;    // target lane group
            int tj  = (kl & 16) ? ((kl & 7) + 8) : (kl & 7);
#pragma unroll
            for (int r = 0; r < 8; ++r) {
                int m = r + c8;             // row within tile
                st[(size_t)(kt2 * 32 + m + hi) * 16 + tj] = (__bf16)silu_f(acc[nt][r]);
            }
        }
        // Wave-local cross-lane LDS dependency: wait for DS writes to land.
        asm volatile("s_wait_dscnt 0" ::: "memory");

        // --- second GEMM: silu(pre) @ W_msg ---
        v8f acc2[8];
#pragma unroll
        for (int nt = 0; nt < 8; ++nt)
#pragma unroll
            for (int r = 0; r < 8; ++r) acc2[nt][r] = 0.0f;
#pragma unroll
        for (int kt = 0; kt < 4; ++kt) {
            v16bf a2 = *(const v16bf*)(st + (size_t)kt * FRAG + (size_t)lane * 16);
#pragma unroll
            for (int nt = 0; nt < 8; ++nt)
                acc2[nt] = wmma_bf16(a2, ldfrag(sWmsg, kt, nt, lane), acc2[nt]);
        }

        // --- cutoff scale + scatter-add to agg[dst] ---
        float4 cva = *(const float4*)(Cenv + e0 + c8);
        float4 cvb = *(const float4*)(Cenv + e0 + c8 + 4);
        int4   dra = *(const int4*)(dstv + e0 + c8);
        int4   drb = *(const int4*)(dstv + e0 + c8 + 4);
        float cv[8] = {cva.x, cva.y, cva.z, cva.w, cvb.x, cvb.y, cvb.z, cvb.w};
        int drow[8] = {dra.x, dra.y, dra.z, dra.w, drb.x, drb.y, drb.z, drb.w};
#pragma unroll
        for (int nt = 0; nt < 8; ++nt) {
            int col = nt * 16 + m16;
#pragma unroll
            for (int r = 0; r < 8; ++r)
                atomicAdd(agg + (size_t)drow[r] * Dd + col, acc2[nt][r] * cv[r]);
        }
    }
}

// ---------------------------------------------------------------------------
// Node kernel: h_out = h + silu(h*Wself + agg*Wagg + b_node)
// ---------------------------------------------------------------------------
__global__ __launch_bounds__(128)
void node_kernel(const float* __restrict__ h,
                 const float* __restrict__ agg,
                 const float* __restrict__ bnode,
                 const __bf16* __restrict__ pWself,
                 const __bf16* __restrict__ pWagg,
                 float* __restrict__ out) {
    const int lane = threadIdx.x & 31;
    const int wave = threadIdx.x >> 5;
    const int m16  = lane & 15;
    const int c8   = (lane & 16) ? 8 : 0;
    const int ntile = Nn / 16;   // 3125, exact

    for (int t = blockIdx.x * 4 + wave; t < ntile; t += gridDim.x * 4) {
        const int n0 = t * 16;
        const float* rh = h   + (size_t)(n0 + m16) * Dd;
        const float* ra = agg + (size_t)(n0 + m16) * Dd;

        v8f acc[8];
#pragma unroll
        for (int nt = 0; nt < 8; ++nt) {
            float b = bnode[nt * 16 + m16];
#pragma unroll
            for (int r = 0; r < 8; ++r) acc[nt][r] = b;
        }
#pragma unroll
        for (int kt = 0; kt < 4; ++kt) {
            v16bf ah = load_a(rh, kt * 32, lane);
            v16bf aa = load_a(ra, kt * 32, lane);
#pragma unroll
            for (int nt = 0; nt < 8; ++nt) {
                acc[nt] = wmma_bf16(ah, ldfrag(pWself, kt, nt, lane), acc[nt]);
                acc[nt] = wmma_bf16(aa, ldfrag(pWagg,  kt, nt, lane), acc[nt]);
            }
        }
#pragma unroll
        for (int nt = 0; nt < 8; ++nt) {
            int col = nt * 16 + m16;
#pragma unroll
            for (int r = 0; r < 8; ++r) {
                int m = r + c8;
                size_t ix = (size_t)(n0 + m) * Dd + col;
                out[ix] = h[ix] + silu_f(acc[nt][r]);
            }
        }
    }
}

// ---------------------------------------------------------------------------
extern "C" void kernel_launch(void* const* d_in, const int* in_sizes, int n_in,
                              void* d_out, int out_size, void* d_ws, size_t ws_size,
                              hipStream_t stream) {
    const float* node_feats = (const float*)d_in[0];
    const float* edge_feats = (const float*)d_in[1];
    const int*   edge_index = (const int*)d_in[2];
    const float* dist       = (const float*)d_in[3];
    const float* W_src      = (const float*)d_in[4];
    const float* W_dst      = (const float*)d_in[5];
    const float* W_edge     = (const float*)d_in[6];
    const float* b_edge     = (const float*)d_in[7];
    const float* W_msg      = (const float*)d_in[8];
    const float* W_self     = (const float*)d_in[9];
    const float* W_agg      = (const float*)d_in[10];
    const float* b_node     = (const float*)d_in[11];
    float* out = (float*)d_out;

    const int* src  = edge_index;
    const int* dstv = edge_index + Ee;

    // Workspace layout (~80 MB): Cenv, h1, h2, agg, packed bf16 weights
    char* ws = (char*)d_ws;
    size_t o = 0;
    auto take = [&](size_t bytes) -> void* {
        void* p = ws + o;
        o += (bytes + 255) & ~(size_t)255;
        return p;
    };
    float*  Cenv   = (float*) take((size_t)Ee * 4);
    float*  h1     = (float*) take((size_t)Nn * Dd * 4);
    float*  h2     = (float*) take((size_t)Nn * Dd * 4);
    float*  agg    = (float*) take((size_t)Nn * Dd * 4);
    __bf16* packed = (__bf16*)take((size_t)Ll * LAYER_HALFS * 2);

    {
        int tot = Ll * LAYER_HALFS;
        pack_kernel<<<(tot + 255) / 256, 256, 0, stream>>>(
            W_src, W_dst, W_msg, W_self, W_agg, W_edge, packed);
    }
    cutoff_kernel<<<(Ee + 255) / 256, 256, 0, stream>>>(dist, Cenv);

    for (int l = 0; l < Ll; ++l) {
        const float* hin  = (l == 0) ? node_feats : ((l == 1) ? h1 : h2);
        float*       hout = (l == 0) ? h1 : ((l == 1) ? h2 : out);
        const __bf16* pl  = packed + (size_t)l * LAYER_HALFS;

        zero_kernel<<<(Nn * Dd + 255) / 256, 256, 0, stream>>>(agg, Nn * Dd);
        edge_kernel<<<2048, 128, 0, stream>>>(
            hin, edge_feats, src, dstv, Cenv, b_edge + (size_t)l * Dd,
            pl + 0 * MAT128, pl + 1 * MAT128, pl + 2 * MAT128, pl + 5 * MAT128, agg);
        node_kernel<<<1024, 128, 0, stream>>>(
            hin, agg, b_node + (size_t)l * Dd,
            pl + 3 * MAT128, pl + 4 * MAT128, hout);
    }
}
